// LSTMParaClassifier_39883066311273
// MI455X (gfx1250) — matrix operand order repaired and checked by reading, not compiled, not run on hardware
//
#include <hip/hip_runtime.h>
#include <hip/hip_bf16.h>

// ---------------------------------------------------------------------------
// 2-layer LSTM (B=64,T=256,IN=512,D=1024) + Dense(16)->Dense(1) head.
// bf16 WMMA (v_wmma_f32_16x16x32_bf16), f32 accumulation, fused L1/L2 steps.
// ---------------------------------------------------------------------------

typedef __bf16 bf16;
typedef __attribute__((ext_vector_type(8)))  __bf16 v8bf;
typedef __attribute__((ext_vector_type(16))) __bf16 v16bf;
typedef __attribute__((ext_vector_type(8)))  float  v8f;

#define BATCH   64
#define TSTEPS  256
#define INDIM   512
#define DHID    1024
#define NGATE   4096   // 4*DHID
#define MROWS   16384  // BATCH*TSTEPS

// ---------------- WMMA helpers (wave32) ----------------

static __device__ inline v8f wmma_bf16(v16bf a, v16bf b, v8f c) {
    return __builtin_amdgcn_wmma_f32_16x16x32_bf16(
        /*neg_a=*/false, a, /*neg_b=*/false, b,
        /*c_mod=*/(short)0, c, /*reuse_a=*/false, /*reuse_b=*/false);
}

// A fragment: 16x32 bf16 tile from row-major [.., ld] matrix.
// lane<16: row m0+lane, K = {k0..k0+15}
// lane>=16: row m0+lane-16, K = {k0+16..k0+31}
static __device__ inline v16bf load_a_frag(const bf16* base, int ld,
                                           int m0, int k0, int lane) {
    int row = m0 + (lane & 15);
    int kb  = k0 + ((lane >> 4) << 4);          // +16 for upper half-wave
    const bf16* p = base + (size_t)row * ld + kb;
    v8bf lo = *reinterpret_cast<const v8bf*>(p);
    v8bf hi = *reinterpret_cast<const v8bf*>(p + 8);
    return __builtin_shufflevector(lo, hi, 0,1,2,3,4,5,6,7,8,9,10,11,12,13,14,15);
}

// B fragment from pre-swizzled buffer: [nTile][kTile][lane][16 bf16]
static __device__ inline const bf16* b_frag_ptr(const bf16* packed, int KT,
                                                int nTile, int kTile, int lane) {
    return packed + ((((size_t)nTile * KT + kTile) * 32 + lane) << 4);
}
static __device__ inline v16bf load_b_frag(const bf16* packed, int KT,
                                           int nTile, int kTile, int lane) {
    return *reinterpret_cast<const v16bf*>(b_frag_ptr(packed, KT, nTile, kTile, lane));
}

static __device__ inline float sigmoidf_(float x) {
    return 1.0f / (1.0f + __expf(-x));
}

// ---------------- elementwise conversion / packing ----------------

__global__ void cvt_f32_bf16_kernel(const float* __restrict__ in,
                                    bf16* __restrict__ out, size_t n) {
    size_t i = (size_t)blockIdx.x * blockDim.x + threadIdx.x;
    if (i < n) out[i] = (bf16)in[i];
}

// Pack row-major fp32 weight W[K][N] into WMMA B-fragment layout (bf16).
__global__ void pack_b_kernel(const float* __restrict__ W,
                              bf16* __restrict__ P, int K, int N) {
    size_t idx = (size_t)blockIdx.x * blockDim.x + threadIdx.x;
    size_t total = (size_t)K * N;
    if (idx >= total) return;
    int k = (int)(idx / N);
    int n = (int)(idx % N);
    int nT = n >> 4, kT = k >> 5, kr = k & 31;
    int lane = (n & 15) + (kr & 16);
    int e    = kr & 15;
    int KT   = K >> 5;
    P[(((size_t)nT * KT + kT) * 32 + lane) * 16 + e] = (bf16)W[idx];
}

// ---------------- big GEMM: Z1pre = X @ k1 + b1 ----------------
// Xb: bf16 [MROWS, INDIM] (row = b*T + t).  Z1: f32 [T][B][NGATE].
// grid = (NGATE/128, MROWS/64), block = 128 (4 waves, each 32M x 64N = 8 accs).
__global__ __launch_bounds__(128, 1)
void gemm_xk1_kernel(const bf16* __restrict__ Xb,
                     const bf16* __restrict__ k1p,
                     const float* __restrict__ b1,
                     float* __restrict__ Z1) {
    const int lane = threadIdx.x & 31;
    const int wave = threadIdx.x >> 5;
    const int m0 = blockIdx.y * 64 + (wave & 1) * 32;
    const int n0 = blockIdx.x * 128 + (wave >> 1) * 64;
    const int KT = INDIM / 32;  // 16

    v8f acc[4][2] = {};  // [tn][tm]
    for (int kt = 0; kt < KT; ++kt) {
        v16bf a0 = load_a_frag(Xb, INDIM, m0, kt * 32, lane);
        v16bf a1 = load_a_frag(Xb, INDIM, m0 + 16, kt * 32, lane);
        if (kt + 1 < KT)  // stream next B fragments through L2
            __builtin_prefetch(b_frag_ptr(k1p, KT, n0 >> 4, kt + 1, lane), 0, 1);
#pragma unroll
        for (int tn = 0; tn < 4; ++tn) {
            v16bf b = load_b_frag(k1p, KT, (n0 >> 4) + tn, kt, lane);
            acc[tn][0] = wmma_bf16(a0, b, acc[tn][0]);
            acc[tn][1] = wmma_bf16(a1, b, acc[tn][1]);
        }
    }
#pragma unroll
    for (int tn = 0; tn < 4; ++tn) {
        int n = n0 + tn * 16 + (lane & 15);
        float bias = b1[n];
#pragma unroll
        for (int tm = 0; tm < 2; ++tm) {
#pragma unroll
            for (int r = 0; r < 8; ++r) {
                int m = m0 + tm * 16 + r + ((lane >> 4) << 3);  // row = b*T + t
                int bb = m >> 8;
                int tt = m & 255;
                Z1[((size_t)(tt * BATCH + bb)) * NGATE + n] = acc[tn][tm][r] + bias;
            }
        }
    }
}

// ---------------- fused recurrent step (software-pipelined layers) ----------
// Launch t (t = 0..TSTEPS):
//   blocks 0..63   : layer-1 step t      (skipped when t == TSTEPS)
//   blocks 64..127 : layer-2 step t-1    (skipped when t == 0)
// Buffer parities: h1_read = h1[t&1] (read by BOTH parts), h1_write = h1[(t+1)&1]
//                  h2_read = h2[(t+1)&1], h2_write = h2[t&1]
// Per block: 2 waves, each 32M x (4 gates x 16N); gate math stays per-lane.
static __device__ inline void mm_accum4x2(v8f acc[4][2],
                                          const bf16* __restrict__ A,
                                          const bf16* __restrict__ Bp,
                                          int j0, int m0, int lane) {
    const int KT = DHID / 32;  // 32
    for (int kt = 0; kt < KT; ++kt) {
        v16bf a0 = load_a_frag(A, DHID, m0, kt * 32, lane);
        v16bf a1 = load_a_frag(A, DHID, m0 + 16, kt * 32, lane);
#pragma unroll
        for (int g = 0; g < 4; ++g) {
            v16bf b = load_b_frag(Bp, KT, g * (DHID / 16) + (j0 >> 4), kt, lane);
            acc[g][0] = wmma_bf16(a0, b, acc[g][0]);
            acc[g][1] = wmma_bf16(a1, b, acc[g][1]);
        }
    }
}

__global__ __launch_bounds__(64, 1)
void lstm_fused_step_kernel(int t,
                            const float* __restrict__ Z1,
                            const bf16* __restrict__ rk1p,
                            const bf16* __restrict__ k2p,
                            const bf16* __restrict__ rk2p,
                            const float* __restrict__ b2,
                            const bf16* __restrict__ h1_read,
                            bf16* __restrict__ h1_write,
                            const bf16* __restrict__ h2_read,
                            bf16* __restrict__ h2_write,
                            float* __restrict__ c1,
                            float* __restrict__ c2,
                            float* __restrict__ h2f) {
    const int lane = threadIdx.x & 31;
    const int wave = threadIdx.x >> 5;   // 0..1
    const int m0 = wave * 32;
    const bool is_l2 = blockIdx.x >= (DHID / 16);
    const int j0 = (is_l2 ? (blockIdx.x - DHID / 16) : blockIdx.x) * 16;
    const int n = j0 + (lane & 15);

    if (!is_l2) {
        // ---------------- layer 1, step t ----------------
        if (t >= TSTEPS) return;
        v8f acc[4][2] = {};
        mm_accum4x2(acc, h1_read, rk1p, j0, m0, lane);
#pragma unroll
        for (int mt = 0; mt < 2; ++mt) {
#pragma unroll
            for (int r = 0; r < 8; ++r) {
                int m = m0 + mt * 16 + r + ((lane >> 4) << 3);
                size_t zr = ((size_t)(t * BATCH + m)) * NGATE;
                float zi = acc[0][mt][r] + Z1[zr + 0 * DHID + n];
                float zf = acc[1][mt][r] + Z1[zr + 1 * DHID + n];
                float zg = acc[2][mt][r] + Z1[zr + 2 * DHID + n];
                float zo = acc[3][mt][r] + Z1[zr + 3 * DHID + n];
                float ig = sigmoidf_(zi);
                float fg = sigmoidf_(zf);
                float gg = tanhf(zg);
                float og = sigmoidf_(zo);
                int ci = m * DHID + n;
                float c = fg * c1[ci] + ig * gg;
                c1[ci] = c;
                h1_write[ci] = (bf16)(og * tanhf(c));
            }
        }
    } else {
        // ---------------- layer 2, step t-1 ----------------
        if (t == 0) return;
        v8f acc[4][2] = {};
        mm_accum4x2(acc, h1_read, k2p, j0, m0, lane);   // h1_{t-1} @ k2
        mm_accum4x2(acc, h2_read, rk2p, j0, m0, lane);  // h2_{t-2} @ rk2
#pragma unroll
        for (int mt = 0; mt < 2; ++mt) {
#pragma unroll
            for (int r = 0; r < 8; ++r) {
                int m = m0 + mt * 16 + r + ((lane >> 4) << 3);
                float zi = acc[0][mt][r] + b2[0 * DHID + n];
                float zf = acc[1][mt][r] + b2[1 * DHID + n];
                float zg = acc[2][mt][r] + b2[2 * DHID + n];
                float zo = acc[3][mt][r] + b2[3 * DHID + n];
                float ig = sigmoidf_(zi);
                float fg = sigmoidf_(zf);
                float gg = tanhf(zg);
                float og = sigmoidf_(zo);
                int ci = m * DHID + n;
                float c = fg * c2[ci] + ig * gg;
                c2[ci] = c;
                float h = og * tanhf(c);
                h2_write[ci] = (bf16)h;
                h2f[ci] = h;
            }
        }
    }
}

// ---------------- classifier head: Dense(16) -> Dense(1) ----------------
// 1024 threads: thread (b,u) computes one 1024-length dot; LDS combine.
__global__ void head_kernel(const float* __restrict__ h2f,
                            const float* __restrict__ dw1,
                            const float* __restrict__ db1,
                            const float* __restrict__ dw2,
                            const float* __restrict__ db2,
                            float* __restrict__ out) {
    __shared__ float part[16][BATCH];
    int tid = threadIdx.x;
    int b = tid & (BATCH - 1);
    int u = tid >> 6;
    float s = db1[u];
    const float* hr = h2f + (size_t)b * DHID;
    for (int j = 0; j < DHID; ++j) s += hr[j] * dw1[j * 16 + u];
    part[u][b] = s * dw2[u];
    __syncthreads();
    if (tid < BATCH) {
        float acc = db2[0];
#pragma unroll
        for (int uu = 0; uu < 16; ++uu) acc += part[uu][tid];
        out[tid] = acc;
    }
}

// ---------------------------------------------------------------------------

static inline size_t align_up(size_t x, size_t a) { return (x + a - 1) & ~(a - 1); }

extern "C" void kernel_launch(void* const* d_in, const int* in_sizes, int n_in,
                              void* d_out, int out_size, void* d_ws, size_t ws_size,
                              hipStream_t stream) {
    (void)in_sizes; (void)n_in; (void)out_size; (void)ws_size;

    const float* x   = (const float*)d_in[0];
    const float* k1  = (const float*)d_in[1];
    const float* rk1 = (const float*)d_in[2];
    const float* b1  = (const float*)d_in[3];
    const float* k2  = (const float*)d_in[4];
    const float* rk2 = (const float*)d_in[5];
    const float* b2  = (const float*)d_in[6];
    const float* dw1 = (const float*)d_in[7];
    const float* db1 = (const float*)d_in[8];
    const float* dw2 = (const float*)d_in[9];
    const float* db2 = (const float*)d_in[10];
    float* out = (float*)d_out;

    // --- workspace carve-up (~316 MB) ---
    char* base = (char*)d_ws;
    size_t off = 0;
    bf16* Xb    = (bf16*)(base + off); off = align_up(off + (size_t)MROWS * INDIM * 2, 256);
    bf16* k1p   = (bf16*)(base + off); off = align_up(off + (size_t)INDIM * NGATE * 2, 256);
    bf16* rk1p  = (bf16*)(base + off); off = align_up(off + (size_t)DHID * NGATE * 2, 256);
    bf16* k2p   = (bf16*)(base + off); off = align_up(off + (size_t)DHID * NGATE * 2, 256);
    bf16* rk2p  = (bf16*)(base + off); off = align_up(off + (size_t)DHID * NGATE * 2, 256);
    float* Z1   = (float*)(base + off); off = align_up(off + (size_t)MROWS * NGATE * 4, 256);
    bf16* h1b   = (bf16*)(base + off); off = align_up(off + (size_t)2 * BATCH * DHID * 2, 256);
    bf16* h2b   = (bf16*)(base + off); off = align_up(off + (size_t)2 * BATCH * DHID * 2, 256);
    float* c1   = (float*)(base + off); off = align_up(off + (size_t)BATCH * DHID * 4, 256);
    float* c2   = (float*)(base + off); off = align_up(off + (size_t)BATCH * DHID * 4, 256);
    float* h2f  = (float*)(base + off); off = align_up(off + (size_t)BATCH * DHID * 4, 256);

    // --- zero recurrent state (capture-safe async memset) ---
    hipMemsetAsync(h1b, 0, (size_t)2 * BATCH * DHID * 2, stream);
    hipMemsetAsync(h2b, 0, (size_t)2 * BATCH * DHID * 2, stream);
    hipMemsetAsync(c1, 0, (size_t)BATCH * DHID * 4, stream);
    hipMemsetAsync(c2, 0, (size_t)BATCH * DHID * 4, stream);

    // --- convert X to bf16 ---
    {
        size_t n = (size_t)MROWS * INDIM;
        cvt_f32_bf16_kernel<<<(unsigned)((n + 255) / 256), 256, 0, stream>>>(x, Xb, n);
    }
    // --- pack weights into B-fragment layout ---
    {
        size_t n = (size_t)INDIM * NGATE;
        pack_b_kernel<<<(unsigned)((n + 255) / 256), 256, 0, stream>>>(k1, k1p, INDIM, NGATE);
        n = (size_t)DHID * NGATE;
        pack_b_kernel<<<(unsigned)((n + 255) / 256), 256, 0, stream>>>(rk1, rk1p, DHID, NGATE);
        pack_b_kernel<<<(unsigned)((n + 255) / 256), 256, 0, stream>>>(k2,  k2p,  DHID, NGATE);
        pack_b_kernel<<<(unsigned)((n + 255) / 256), 256, 0, stream>>>(rk2, rk2p, DHID, NGATE);
    }

    // --- hoisted input projection: Z1pre = X@k1 + b1 ---
    {
        dim3 grid(NGATE / 128, MROWS / 64);  // (32, 256)
        gemm_xk1_kernel<<<grid, 128, 0, stream>>>(Xb, k1p, b1, Z1);
    }

    // --- sequential recurrence, two layers pipelined per launch ---
    const size_t HS = (size_t)BATCH * DHID;
    for (int t = 0; t <= TSTEPS; ++t) {
        bf16* h1_read  = h1b + (size_t)(t & 1) * HS;
        bf16* h1_write = h1b + (size_t)((t + 1) & 1) * HS;
        bf16* h2_read  = h2b + (size_t)((t + 1) & 1) * HS;
        bf16* h2_write = h2b + (size_t)(t & 1) * HS;
        lstm_fused_step_kernel<<<2 * (DHID / 16), 64, 0, stream>>>(
            t, Z1, rk1p, k2p, rk2p, b2,
            h1_read, h1_write, h2_read, h2_write, c1, c2, h2f);
    }

    // --- head ---
    head_kernel<<<1, 1024, 0, stream>>>(h2f, dw1, db1, dw2, db2, out);
}
